// FuzzyTM_13022340842104
// MI455X (gfx1250) — matrix operand order repaired and checked by reading, not compiled, not run on hardware
//
#include <hip/hip_runtime.h>
#include <stdint.h>

#define N_STATES 4096
#define SEQ_LEN  2048
#define KTILES   128     // 4096 / 32
#define JTILES   256     // 4096 / 16

typedef __attribute__((ext_vector_type(16))) __bf16 v16bf;
typedef __attribute__((ext_vector_type(8)))  float  v8f;

__device__ __forceinline__ unsigned short f2bf(float f) {
    // round-to-nearest (ties away) f32 -> bf16; inputs are normalized probs (>=0)
    unsigned u = __float_as_uint(f);
    return (unsigned short)((u + 0x8000u) >> 16);
}

// ---------------------------------------------------------------------------
// Softmax each row of T ([4096,2,4096]) and scatter into WMMA B-fragment
// layout: Tpack[sym][frag = jtile*128+ktile][lane][16 ushort], where for
// element (k=i%32, n=j%16): lane = n + 16*(k>=16), dword v=(k%16)/2, slot=k%2.
// ---------------------------------------------------------------------------
__global__ void __launch_bounds__(256)
softmax_pack_T(const float* __restrict__ T, unsigned short* __restrict__ Tpack) {
    __shared__ float red[256];
    const int row = blockIdx.x;           // 0..8191  (= i*2 + sym)
    const int i   = row >> 1;
    const int sym = row & 1;
    const int tid = threadIdx.x;
    const float* src = T + (size_t)row * N_STATES;

    float v[16];
    float m = -3.4e38f;
#pragma unroll
    for (int c = 0; c < 16; ++c) {
        v[c] = src[tid + 256 * c];
        m = fmaxf(m, v[c]);
    }
    red[tid] = m; __syncthreads();
    for (int s = 128; s > 0; s >>= 1) {
        if (tid < s) red[tid] = fmaxf(red[tid], red[tid + s]);
        __syncthreads();
    }
    m = red[0]; __syncthreads();

    float sum = 0.f;
#pragma unroll
    for (int c = 0; c < 16; ++c) { v[c] = __expf(v[c] - m); sum += v[c]; }
    red[tid] = sum; __syncthreads();
    for (int s = 128; s > 0; s >>= 1) {
        if (tid < s) red[tid] += red[tid + s];
        __syncthreads();
    }
    const float inv = 1.0f / red[0];

    const int ktile   = i >> 5;
    const int kr      = i & 31;
    const int lane_hi = (kr & 16) ? 16 : 0;          // lanes 16..31 hold K 16..31
    const int vo      = ((kr & 15) >> 1) * 2 + (kr & 1);  // ushort offset in lane's 32B
    unsigned short* base = Tpack + ((size_t)sym << 24);
#pragma unroll
    for (int c = 0; c < 16; ++c) {
        const int j = tid + 256 * c;
        const int jtile = j >> 4, n = j & 15;
        const size_t off = ((size_t)(jtile * KTILES + ktile) << 9)
                         + (size_t)(n + lane_hi) * 16 + vo;
        base[off] = f2bf(v[c] * inv);
    }
}

// Softmax of O rows ([4096,2,4]) -> f32 On[(i*2+sym)*4 + r]
__global__ void __launch_bounds__(256)
softmax_O(const float* __restrict__ O, float* __restrict__ On) {
    const int r = blockIdx.x * 256 + threadIdx.x;    // 0..8191
    if (r >= N_STATES * 2) return;
    float a = O[r * 4 + 0], b = O[r * 4 + 1], c = O[r * 4 + 2], d = O[r * 4 + 3];
    float m = fmaxf(fmaxf(a, b), fmaxf(c, d));
    a = __expf(a - m); b = __expf(b - m); c = __expf(c - m); d = __expf(d - m);
    const float inv = 1.f / (a + b + c + d);
    On[r * 4 + 0] = a * inv; On[r * 4 + 1] = b * inv;
    On[r * 4 + 2] = c * inv; On[r * 4 + 3] = d * inv;
}

// s0 = softmax(init) -> partial slice 0 of state buffer A; slices 1..3 zeroed
__global__ void __launch_bounds__(256)
init_state(const float* __restrict__ ini, float* __restrict__ sA) {
    __shared__ float red[256];
    const int tid = threadIdx.x;
    float v[16];
    float m = -3.4e38f;
#pragma unroll
    for (int c = 0; c < 16; ++c) { v[c] = ini[tid + 256 * c]; m = fmaxf(m, v[c]); }
    red[tid] = m; __syncthreads();
    for (int s = 128; s > 0; s >>= 1) {
        if (tid < s) red[tid] = fmaxf(red[tid], red[tid + s]);
        __syncthreads();
    }
    m = red[0]; __syncthreads();
    float sum = 0.f;
#pragma unroll
    for (int c = 0; c < 16; ++c) { v[c] = __expf(v[c] - m); sum += v[c]; }
    red[tid] = sum; __syncthreads();
    for (int s = 128; s > 0; s >>= 1) {
        if (tid < s) red[tid] += red[tid + s];
        __syncthreads();
    }
    const float inv = 1.0f / red[0];
#pragma unroll
    for (int c = 0; c < 16; ++c) {
        const int j = tid + 256 * c;
        sA[j] = v[c] * inv;
        sA[1 * N_STATES + j] = 0.f;
        sA[2 * N_STATES + j] = 0.f;
        sA[3 * N_STATES + j] = 0.f;
    }
}

// ---------------------------------------------------------------------------
// One scan step. Blocks 0..127: WMMA matvec (block = one K-slice of 1024,
// 8 waves = 8 jtiles). Block 128: emission dot -> out[t*4..t*4+3].
// State = 4 f32 partial vectors (sum over K-slices) to avoid atomics.
// ---------------------------------------------------------------------------
__global__ void __launch_bounds__(256)
step_kernel(const int* __restrict__ seq, int t,
            const unsigned short* __restrict__ Tpack,
            const float* __restrict__ On,
            const float* __restrict__ s_in,    // [4][4096] f32 partials
            float* __restrict__ s_out,         // [4][4096] f32 partials
            float* __restrict__ out)           // [SEQ_LEN*4]
{
    __shared__ unsigned short abf[1024];       // bf16 state chunk for this K-slice
    __shared__ float red[1024];                // emission reduction
    const int tid = threadIdx.x;
    const int sym = seq[t];

    if (blockIdx.x < 128) {
        const int kslice = blockIdx.x & 3;
        const int jbase  = (blockIdx.x >> 2) * 8;

        // Stage summed state (4 partials) for this K-slice into LDS as bf16.
#pragma unroll
        for (int c = 0; c < 4; ++c) {
            const int k  = tid + 256 * c;              // 0..1023 local K
            const int kg = kslice * 1024 + k;
            const float s = s_in[kg] + s_in[N_STATES + kg]
                          + s_in[2 * N_STATES + kg] + s_in[3 * N_STATES + kg];
            abf[k] = f2bf(s);
        }
        __syncthreads();

        const int wave  = tid >> 5;
        const int lane  = tid & 31;
        const int jtile = jbase + wave;
        const int koff  = (lane & 16) ? 16 : 0;  // A frag: lanes 0-15 K0..15, 16-31 K16..31
        const unsigned short* tp = Tpack + ((size_t)sym << 24)
            + ((size_t)(jtile * KTILES + kslice * 32) << 9) + (size_t)lane * 16;

        v8f acc = {0.f, 0.f, 0.f, 0.f, 0.f, 0.f, 0.f, 0.f};
#pragma unroll 4
        for (int kt = 0; kt < 32; ++kt) {
            union { uint4 q[2]; v16bf v; } A, B;
            const uint4* ap = (const uint4*)(&abf[kt * 32 + koff]);
            A.q[0] = ap[0];        // K koff..koff+7
            A.q[1] = ap[1];        // K koff+8..koff+15  (+16 bytes)
            const uint4* bp = (const uint4*)(tp + (size_t)kt * 512);
            B.q[0] = bp[0];
            B.q[1] = bp[1];
            acc = __builtin_amdgcn_wmma_f32_16x16x32_bf16(
                false, A.v, false, B.v, (short)0, acc, false, false);
        }
        // All D rows identical (A rows replicated); row 0 lives in VGPR0, lanes 0..15.
        if (lane < 16) {
            s_out[kslice * N_STATES + jtile * 16 + lane] = acc[0];
        }
    } else {
        // Emission: out_prob = s_t @ On[:,sym,:]
        float p0 = 0.f, p1 = 0.f, p2 = 0.f, p3 = 0.f;
        for (int i = tid; i < N_STATES; i += 256) {
            const float s = s_in[i] + s_in[N_STATES + i]
                          + s_in[2 * N_STATES + i] + s_in[3 * N_STATES + i];
            const float* o = On + (size_t)(i * 2 + sym) * 4;
            p0 = fmaf(s, o[0], p0); p1 = fmaf(s, o[1], p1);
            p2 = fmaf(s, o[2], p2); p3 = fmaf(s, o[3], p3);
        }
        red[tid] = p0; red[256 + tid] = p1; red[512 + tid] = p2; red[768 + tid] = p3;
        __syncthreads();
        for (int s = 128; s > 0; s >>= 1) {
            if (tid < s) {
                red[tid]       += red[tid + s];
                red[256 + tid] += red[256 + tid + s];
                red[512 + tid] += red[512 + tid + s];
                red[768 + tid] += red[768 + tid + s];
            }
            __syncthreads();
        }
        if (tid == 0) {
            out[t * 4 + 0] = red[0];
            out[t * 4 + 1] = red[256];
            out[t * 4 + 2] = red[512];
            out[t * 4 + 3] = red[768];
        }
    }
}

extern "C" void kernel_launch(void* const* d_in, const int* in_sizes, int n_in,
                              void* d_out, int out_size, void* d_ws, size_t ws_size,
                              hipStream_t stream) {
    const int*   seq  = (const int*)d_in[0];
    const float* T    = (const float*)d_in[1];
    const float* O    = (const float*)d_in[2];
    const float* ini  = (const float*)d_in[3];
    float* out = (float*)d_out;

    char* ws = (char*)d_ws;
    unsigned short* Tpack = (unsigned short*)ws;                       // 64 MiB (2 * 2^24 bf16)
    float* On = (float*)(ws + 67108864);                               // 128 KiB
    float* sA = (float*)(ws + 67108864 + 131072);                      // 64 KiB (4 partials)
    float* sB = (float*)(ws + 67108864 + 131072 + 65536);              // 64 KiB

    softmax_pack_T<<<8192, 256, 0, stream>>>(T, Tpack);
    softmax_O<<<32, 256, 0, stream>>>(O, On);
    init_state<<<1, 256, 0, stream>>>(ini, sA);

    for (int t = 0; t < SEQ_LEN; ++t) {
        float* si = (t & 1) ? sB : sA;
        float* so = (t & 1) ? sA : sB;
        step_kernel<<<129, 256, 0, stream>>>(seq, t, Tpack, On, si, so, out);
    }
}
